// SimpleGRU_17746804867097
// MI455X (gfx1250) — matrix operand order repaired and checked by reading, not compiled, not run on hardware
//
#include <hip/hip_runtime.h>
#include <math.h>

#define VOCAB 50257
#define EMB   256
#define HID   256
#define H3    768
#define BATCH 64
#define TLEN  2048

typedef __attribute__((ext_vector_type(16))) __bf16 v16bf;
typedef __attribute__((ext_vector_type(8)))  __bf16 v8bf;
typedef __attribute__((ext_vector_type(4)))  __bf16 v4bf;
typedef __attribute__((ext_vector_type(8)))  float  v8f;
typedef __attribute__((ext_vector_type(4)))  unsigned int u32x4;
typedef __attribute__((ext_vector_type(8)))  int          i32x8;
typedef __attribute__((ext_vector_type(4)))  int          i32x4;

// Async LDS<->global builtins take pointer-to-int4 in addrspace(1)/(3)
// (exact types recovered from the round-3 diagnostics).
typedef __attribute__((address_space(1))) i32x4 as1_v4i;
typedef __attribute__((address_space(3))) i32x4 as3_v4i;
#define GPTR(p) ((as1_v4i*)(unsigned long long)(p))
#define LPTR(p) ((as3_v4i*)(unsigned)(unsigned long long)(p))

#if __has_builtin(__builtin_amdgcn_tensor_load_to_lds)
#define HAVE_TDM 1
#endif
#if __has_builtin(__builtin_amdgcn_global_load_async_to_lds_b128)
#define HAVE_ASYNC_LD 1
#endif
#if __has_builtin(__builtin_amdgcn_global_store_async_from_lds_b128)
#define HAVE_ASYNC_ST 1
#endif

__device__ __forceinline__ float sigmoidf_(float x) {
  return 1.0f / (1.0f + __expf(-x));
}

__device__ __forceinline__ v16bf frag_from_halves(v8bf lo, v8bf hi) {
  return __builtin_shufflevector(lo, hi, 0, 1, 2, 3, 4, 5, 6, 7,
                                 8, 9, 10, 11, 12, 13, 14, 15);
}

// Wait for all outstanding async LDS<->global transfers of this wave.
__device__ __forceinline__ void wait_async_all() {
#if defined(HAVE_ASYNC_LD) || defined(HAVE_ASYNC_ST)
#if __has_builtin(__builtin_amdgcn_s_wait_asynccnt)
  __builtin_amdgcn_s_wait_asynccnt(0);
#else
  asm volatile("s_wait_asynccnt 0x0" ::: "memory");
#endif
#endif
}

#ifdef HAVE_TDM
// TDM: DMA a 2-D tile (128 rows x 256 elems x 2B, row stride 256) -> LDS.
// Call from a single wave (EXEC ignored by TDM; wave-uniform branch gates issue).
__device__ __forceinline__ void tdm_load_tile_128x256(const __bf16* gsrc,
                                                      __bf16* ldst) {
  unsigned ldsOff = (unsigned)(unsigned long long)ldst;
  unsigned long long ga = (unsigned long long)(const void*)gsrc;
  u32x4 g0 = {
      1u,                                               // count=1, user mode
      ldsOff,                                           // D#.lds_addr
      (unsigned)(ga & 0xFFFFFFFFu),                     // global_addr[31:0]
      (unsigned)((ga >> 32) & 0x01FFFFFFu) | (2u << 30) // addr[56:32]|type=2
  };
  i32x8 g1 = {
      0x00010000,          // workgroup_mask=0, data_size=1 (2 bytes)
      (int)(256u << 16),   // tensor_dim0 = 256 (lo16 into [31:16])
      (int)(128u << 16),   // tensor_dim0 hi=0 | tensor_dim1 = 128
      (int)(256u << 16),   // tensor_dim1 hi=0 | tile_dim0 = 256
      128,                 // tile_dim1 = 128 | tile_dim2 = 0
      256,                 // tensor_dim0_stride lo32 = 256
      0, 0};
  i32x4 gz = {0, 0, 0, 0};
#if defined(__clang_major__) && __clang_major__ >= 23
  i32x8 gz8 = {0, 0, 0, 0, 0, 0, 0, 0};
  __builtin_amdgcn_tensor_load_to_lds(g0, g1, gz, gz, gz8, 0);
#else
  __builtin_amdgcn_tensor_load_to_lds(g0, g1, gz, gz, 0);
#endif
}
#endif

// ---------------------------------------------------------------------------
// Kernel 1a: GRU weights fp32 -> bf16 ([n][k], k contiguous: B-fragment =
// one 32B contiguous load). Kernel 1b: embedding table fp32 -> bf16 (halves
// the gather traffic in kernel 2 and deletes all per-tile cvts there).
// ---------------------------------------------------------------------------
__global__ void gru_prep_w(const float* __restrict__ Wih,
                           const float* __restrict__ Whh,
                           __bf16* __restrict__ WihB,
                           __bf16* __restrict__ WhhB) {
  int i = blockIdx.x * blockDim.x + threadIdx.x;
  if (i < H3 * EMB) {
    WihB[i] = (__bf16)Wih[i];
    WhhB[i] = (__bf16)Whh[i];
  }
}

__global__ void gru_prep_emb(const float* __restrict__ emb,
                             __bf16* __restrict__ embB) {
  long i4 = (long)blockIdx.x * blockDim.x + threadIdx.x;
  if (i4 < (long)VOCAB * EMB / 4) {
    float4 v = ((const float4*)emb)[i4];
    v4bf o = {(__bf16)v.x, (__bf16)v.y, (__bf16)v.z, (__bf16)v.w};
    *(v4bf*)(embB + i4 * 4) = o;
  }
}

// ---------------------------------------------------------------------------
// Kernel 2: GI[t][b][0:768] = emb[tokens[b][t]] @ W_ih^T + b_ih   (bf16 out)
// One block per 64 flattened rows (R = t*64+b); X staged ONCE, then the six
// 128-column W_ih tiles are software-pipelined: TDM DMAs tile nb+1 into the
// alternate LDS buffer while WMMAs consume tile nb. X gather / GI stores use
// async LDS<->global ops when the toolchain exposes them.
// ---------------------------------------------------------------------------
__global__ void __launch_bounds__(256)
gru_gi_wmma(const int*    __restrict__ tokens,
            const __bf16* __restrict__ embB,
            const __bf16* __restrict__ WihB,
            const float*  __restrict__ b_ih,
            __bf16* __restrict__ GI) {
  __shared__ __bf16 sX[64][EMB];        //  32 KB  X tile
  __shared__ __bf16 sW[2][128][EMB];    // 128 KB  double-buffered W tiles
  __shared__ __bf16 stage[64][128];     //  16 KB  store-coalescing buffer

  const int tid  = threadIdx.x;
  const int lane = tid & 31;
  const int wave = tid >> 5;
  const int rowBase = blockIdx.x * 64;   // flattened (t*64+b)

  // ---- Stage X once: bf16 gather, 4 threads/row, 128B each. ----
  {
    int r  = tid >> 2;
    int c0 = (tid & 3) * 64;
    int R  = rowBase + r;
    int t  = R >> 6;
    int b  = R & 63;
    int tok = tokens[b * TLEN + t];
    const __bf16* xsrc = embB + (long)tok * EMB + c0;
    __bf16* xdst = &sX[r][c0];
#ifdef HAVE_ASYNC_LD
    __builtin_amdgcn_global_load_async_to_lds_b128(GPTR(xsrc), LPTR(xdst), 0, 0);
    __builtin_amdgcn_global_load_async_to_lds_b128(GPTR(xsrc), LPTR(xdst), 16, 0);
    __builtin_amdgcn_global_load_async_to_lds_b128(GPTR(xsrc), LPTR(xdst), 32, 0);
    __builtin_amdgcn_global_load_async_to_lds_b128(GPTR(xsrc), LPTR(xdst), 48, 0);
    __builtin_amdgcn_global_load_async_to_lds_b128(GPTR(xsrc), LPTR(xdst), 64, 0);
    __builtin_amdgcn_global_load_async_to_lds_b128(GPTR(xsrc), LPTR(xdst), 80, 0);
    __builtin_amdgcn_global_load_async_to_lds_b128(GPTR(xsrc), LPTR(xdst), 96, 0);
    __builtin_amdgcn_global_load_async_to_lds_b128(GPTR(xsrc), LPTR(xdst), 112, 0);
#else
    const v16bf* s = (const v16bf*)xsrc;
    v16bf* d = (v16bf*)xdst;
    d[0] = s[0]; d[1] = s[1]; d[2] = s[2]; d[3] = s[3];
#endif
  }

#ifdef HAVE_TDM
  if (tid < 32) tdm_load_tile_128x256(WihB, &sW[0][0][0]);   // prologue DMA
#endif

  const int aM   = lane & 15;
  const int nLoc = (wave << 4) + (lane & 15);

#pragma unroll 1
  for (int nb = 0; nb < 6; ++nb) {
    const int buf = nb & 1;
#ifndef HAVE_TDM
    {   // manual W staging fallback
      int n  = tid >> 1;
      int c0 = (tid & 1) * 128;
      const v16bf* s =
          (const v16bf*)(WihB + (long)(nb * 128 + n) * EMB + c0);
      v16bf* d = (v16bf*)&sW[buf][n][c0];
      d[0] = s[0]; d[1] = s[1]; d[2] = s[2]; d[3] = s[3];
      d[4] = s[4]; d[5] = s[5]; d[6] = s[6]; d[7] = s[7];
    }
#else
    if (tid < 32) __builtin_amdgcn_s_wait_tensorcnt(0);  // tile nb landed
#endif
    wait_async_all();   // X gather (nb==0) / previous async stores complete
    __syncthreads();    // sW[buf], sX visible; prev stage readers done
#ifdef HAVE_TDM
    if (nb < 5 && tid < 32)   // overlap: DMA tile nb+1 while computing nb
      tdm_load_tile_128x256(WihB + (long)(nb + 1) * 128 * EMB,
                            &sW[buf ^ 1][0][0]);
#endif

    v8f acc[4];
#pragma unroll
    for (int mt = 0; mt < 4; ++mt)
#pragma unroll
      for (int e = 0; e < 8; ++e) acc[mt][e] = 0.0f;

#pragma unroll
    for (int kc = 0; kc < 8; ++kc) {
      int kb = (kc << 5) + ((lane < 16) ? 0 : 16);
      v16bf bf = *(const v16bf*)&sW[buf][nLoc][kb];   // 32B contiguous LDS
#pragma unroll
      for (int mt = 0; mt < 4; ++mt) {
        int m  = (mt << 4) + aM;
        int k0 = (kc << 5) + ((lane < 16) ? 0 : 16);
        int k1 = (kc << 5) + ((lane < 16) ? 8 : 24);
        v16bf af = frag_from_halves(*(const v8bf*)&sX[m][k0],
                                    *(const v8bf*)&sX[m][k1]);
        acc[mt] = __builtin_amdgcn_wmma_f32_16x16x32_bf16(
            false, af, false, bf, (short)0, acc[mt], false, false);
      }
    }

    // Epilogue: +b_ih, restage to LDS (D layout: N=lane%16, M=v+(lane<16?0:8)).
    const float bias = b_ih[nb * 128 + nLoc];
#pragma unroll
    for (int mt = 0; mt < 4; ++mt) {
#pragma unroll
      for (int v = 0; v < 8; ++v) {
        int mLoc = (mt << 4) + v + ((lane < 16) ? 0 : 8);
        stage[mLoc][nLoc] = (__bf16)(acc[mt][v] + bias);
      }
    }
    __syncthreads();

    // Coalesced 64B-per-thread stores of the 64x128 tile.
    {
      int row = tid >> 2;
      int q   = tid & 3;
      const __bf16* lsrc = &stage[row][q * 32];
      __bf16* gdst = GI + (long)(rowBase + row) * H3 + nb * 128 + q * 32;
#ifdef HAVE_ASYNC_ST
      __builtin_amdgcn_global_store_async_from_lds_b128(GPTR(gdst), LPTR(lsrc), 0, 0);
      __builtin_amdgcn_global_store_async_from_lds_b128(GPTR(gdst), LPTR(lsrc), 16, 0);
      __builtin_amdgcn_global_store_async_from_lds_b128(GPTR(gdst), LPTR(lsrc), 32, 0);
      __builtin_amdgcn_global_store_async_from_lds_b128(GPTR(gdst), LPTR(lsrc), 48, 0);
#else
      const v16bf* s = (const v16bf*)lsrc;
      v16bf* d = (v16bf*)gdst;
      d[0] = s[0]; d[1] = s[1];
#endif
    }
  }
  wait_async_all();   // drain async stores (S_ENDPGM also waits idle)
}

// ---------------------------------------------------------------------------
// Kernel 3: persistent recurrent GRU. Single workgroup, 512 threads (16 waves).
// Per step: GH[64,768] = H[64,256] @ Whh^T via WMMA (wave w owns N-tiles
// 3w..3w+2, all 4 M-tiles -> 96 wmma/wave/step). H state lives in fp32
// REGISTERS; only a bf16 shadow of H is in LDS for the WMMA A-operand.
// LDS: sHb 32KB + sG 192KB = 224KB (<= 320KB per WGP).
// ---------------------------------------------------------------------------
__global__ void __launch_bounds__(512)
gru_recurrent_wmma(const __bf16* __restrict__ GI,
                   const __bf16* __restrict__ WhhB,
                   const float* __restrict__ b_hh,
                   const float* __restrict__ W_fc,
                   const float* __restrict__ b_fc,
                   float* __restrict__ out) {
  __shared__ __bf16 sHb[BATCH][HID];  //  32 KB
  __shared__ float  sG[BATCH][H3];    // 192 KB

  const int tid  = threadIdx.x;
  const int lane = tid & 31;
  const int wave = tid >> 5;

  for (int i = tid; i < BATCH * HID; i += 512) (&sHb[0][0])[i] = (__bf16)0.0f;
  float hreg[32];
#pragma unroll
  for (int i = 0; i < 32; ++i) hreg[i] = 0.0f;

  const int j    = tid & 255;
  const int bTop = tid >> 8;
  const float bhr = b_hh[j];
  const float bhz = b_hh[256 + j];
  const float bhn = b_hh[512 + j];

  const int aM   = lane & 15;
  const int nSel = lane & 15;
  __syncthreads();

  for (int t = 0; t < TLEN; ++t) {
    __builtin_prefetch(&GI[((long)t * BATCH + bTop) * H3 + j], 0, 0);

    v8f acc[4][3];
#pragma unroll
    for (int mt = 0; mt < 4; ++mt)
#pragma unroll
      for (int nt = 0; nt < 3; ++nt)
#pragma unroll
        for (int e = 0; e < 8; ++e) acc[mt][nt][e] = 0.0f;

#pragma unroll 2
    for (int kc = 0; kc < 8; ++kc) {
      v16bf bf[3];
#pragma unroll
      for (int nt = 0; nt < 3; ++nt) {
        int nGlob = ((wave * 3 + nt) << 4) + nSel;
        const __bf16* wp =
            WhhB + (long)nGlob * EMB + (kc << 5) + ((lane < 16) ? 0 : 16);
        bf[nt] = *reinterpret_cast<const v16bf*>(wp);  // 32B contiguous (L2)
      }
#pragma unroll
      for (int mt = 0; mt < 4; ++mt) {
        int m  = (mt << 4) + aM;
        int k0 = (kc << 5) + ((lane < 16) ? 0 : 16);
        int k1 = (kc << 5) + ((lane < 16) ? 8 : 24);
        v16bf af = frag_from_halves(*(const v8bf*)&sHb[m][k0],
                                    *(const v8bf*)&sHb[m][k1]);
#pragma unroll
        for (int nt = 0; nt < 3; ++nt)
          acc[mt][nt] = __builtin_amdgcn_wmma_f32_16x16x32_bf16(
              false, af, false, bf[nt], (short)0, acc[mt][nt], false, false);
      }
    }

#pragma unroll
    for (int mt = 0; mt < 4; ++mt)
#pragma unroll
      for (int nt = 0; nt < 3; ++nt) {
        int nGlob = ((wave * 3 + nt) << 4) + nSel;
#pragma unroll
        for (int v = 0; v < 8; ++v) {
          int m = (mt << 4) + v + ((lane < 16) ? 0 : 8);
          sG[m][nGlob] = acc[mt][nt][v];
        }
      }
    __syncthreads();

    const long gRow = (long)t * BATCH;
#pragma unroll 4
    for (int i = 0; i < 32; ++i) {
      int b = (i << 1) | bTop;
      long g = (gRow + b) * H3;
      float gir = (float)GI[g + j];
      float giz = (float)GI[g + 256 + j];
      float gin = (float)GI[g + 512 + j];
      float r = sigmoidf_(gir + sG[b][j] + bhr);
      float z = sigmoidf_(giz + sG[b][256 + j] + bhz);
      float n = tanhf(gin + r * (sG[b][512 + j] + bhn));
      float hn = (1.0f - z) * n + z * hreg[i];
      hreg[i] = hn;
      sHb[b][j] = (__bf16)hn;
    }
    __syncthreads();
  }

#pragma unroll
  for (int i = 0; i < 32; ++i) {
    int b = (i << 1) | bTop;
    sG[b][j] = hreg[i];
  }
  __syncthreads();
  if (tid < BATCH * 2) {
    int b = tid >> 1, o = tid & 1;
    float s = b_fc[o];
    const float* w = W_fc + o * HID;
    for (int k = 0; k < HID; ++k) s += sG[b][k] * w[k];
    out[b * 2 + o] = sigmoidf_(s);
  }
}

// ---------------------------------------------------------------------------
extern "C" void kernel_launch(void* const* d_in, const int* in_sizes, int n_in,
                              void* d_out, int out_size, void* d_ws, size_t ws_size,
                              hipStream_t stream) {
  const int*   tokens = (const int*)  d_in[0];
  const float* emb    = (const float*)d_in[1];
  const float* W_ih   = (const float*)d_in[2];
  const float* W_hh   = (const float*)d_in[3];
  const float* b_ih   = (const float*)d_in[4];
  const float* b_hh   = (const float*)d_in[5];
  const float* W_fc   = (const float*)d_in[6];
  const float* b_fc   = (const float*)d_in[7];
  float* out = (float*)d_out;

  char* ws = (char*)d_ws;
  __bf16* WihB = (__bf16*)(ws);                  // 768*256*2   = 384 KB
  __bf16* WhhB = (__bf16*)(ws + 393216);         // 384 KB
  __bf16* embB = (__bf16*)(ws + 786432);         // 50257*256*2 = 24.6 MB
  __bf16* GI   = (__bf16*)(ws + 26518016);       // 131072*768*2 = 192 MB

  gru_prep_w<<<dim3((H3 * EMB + 255) / 256), dim3(256), 0, stream>>>(
      W_ih, W_hh, WihB, WhhB);
  gru_prep_emb<<<dim3((VOCAB * EMB / 4 + 255) / 256), dim3(256), 0, stream>>>(
      emb, embB);

  gru_gi_wmma<<<dim3((BATCH * TLEN) / 64), dim3(256), 0, stream>>>(
      tokens, embB, WihB, b_ih, GI);

  gru_recurrent_wmma<<<dim3(1), dim3(512), 0, stream>>>(
      GI, WhhB, b_hh, W_fc, b_fc, out);
}
